// AttentionHead_51453708206582
// MI455X (gfx1250) — matrix operand order repaired and checked by reading, not compiled, not run on hardware
//
#include <hip/hip_runtime.h>
#include <hip/hip_bf16.h>

typedef _Float16 h16;
typedef __attribute__((ext_vector_type(16))) _Float16 v16h;
typedef __attribute__((ext_vector_type(8)))  _Float16 v8h;
typedef __attribute__((ext_vector_type(8)))  float    v8f;
typedef __attribute__((ext_vector_type(4)))  int      v4i;

#define EMBED 1024
#define DH    128
#define SEQ   2048
#define BATCH 16
#define BS    (BATCH * SEQ)

// ---- gfx1250 async global->LDS copy (ASYNCcnt path), with safe fallback ----
#if defined(__gfx1250__) && __has_builtin(__builtin_amdgcn_global_load_async_to_lds_b128)
#define HAVE_ASYNC_LDS 1
typedef __attribute__((address_space(1))) v4i* g_v4i_ptr;   // global int4*
typedef __attribute__((address_space(3))) v4i* l_v4i_ptr;   // LDS int4*
__device__ __forceinline__ void async_cp16(const void* gsrc, void* lds_dst) {
  __builtin_amdgcn_global_load_async_to_lds_b128(
      (g_v4i_ptr)(uintptr_t)gsrc,
      (l_v4i_ptr)(uint32_t)(uintptr_t)lds_dst,  // low 32 bits of generic = LDS offset
      0, 0);
}
__device__ __forceinline__ void wait_async0() {
#if __has_builtin(__builtin_amdgcn_s_wait_asynccnt)
  __builtin_amdgcn_s_wait_asynccnt(0);
#else
  asm volatile("s_wait_asynccnt 0x0" ::: "memory");
#endif
}
#else
#define HAVE_ASYNC_LDS 0
__device__ __forceinline__ void async_cp16(const void* gsrc, void* lds_dst) {
  *(v8h*)lds_dst = *(const v8h*)gsrc;
}
__device__ __forceinline__ void wait_async0() {}
#endif

// ---- WMMA operand gathers (contiguous k order, same for A and B) ----
// A 16x32 f16: lane l -> row (l&15); slots 0..15 = K kb+0..15, kb = 16*(l>>4). Contiguous.
__device__ __forceinline__ v16h make_a(const h16* row, int kb) {
  v8h lo = *(const v8h*)(row + kb);
  v8h hi = *(const v8h*)(row + kb + 8);
  v16h r;
#pragma unroll
  for (int i = 0; i < 8; ++i) { r[i] = lo[i]; r[i + 8] = hi[i]; }
  return r;
}
// B 32x16 f16: lane l -> col (l&15); slots 0..15 = K kb+0..15, kb = 16*(l>>4). Contiguous.
__device__ __forceinline__ v16h make_b(const h16* row, int kb) {
  v8h lo = *(const v8h*)(row + kb);
  v8h hi = *(const v8h*)(row + kb + 8);
  v16h r;
#pragma unroll
  for (int i = 0; i < 8; ++i) { r[i] = lo[i]; r[i + 8] = hi[i]; }
  return r;
}

__device__ __forceinline__ v8f wmma_f16(v16h a, v16h b, v8f c) {
  return __builtin_amdgcn_wmma_f32_16x16x32_f16(false, a, false, b, (short)0, c,
                                                false, false);
}

// ============ Kernel 1: fused QKV projection, fp32 -> f16 out ============
__global__ __launch_bounds__(256) void qkv_proj_kernel(
    const float* __restrict__ x,
    const float* __restrict__ Wq, const float* __restrict__ bq,
    const float* __restrict__ Wk, const float* __restrict__ bk,
    const float* __restrict__ Wv, const float* __restrict__ bv,
    h16* __restrict__ qkv_ws) {
  __shared__ h16 Xs[128][40];   // 128 rows x 32 K (padded)
  __shared__ h16 Wt[128][40];   // transposed: [n][k]

  const int which = blockIdx.y;
  const float* W   = (which == 0) ? Wq : (which == 1) ? Wk : Wv;
  const float* bv_ = (which == 0) ? bq : (which == 1) ? bk : bv;
  h16* out = qkv_ws + (size_t)which * BS * DH;

  const int r0   = blockIdx.x * 128;
  const int t    = threadIdx.x;
  const int w    = t >> 5;
  const int lane = t & 31;
  const int h    = lane >> 4;
  const int ln   = lane & 15;

  v8f acc[8];
#pragma unroll
  for (int n = 0; n < 8; ++n) acc[n] = (v8f)(0.0f);

  for (int k0 = 0; k0 < EMBED; k0 += 32) {
    // stage x chunk [128 x 32] as f16
    {
      const int r = t >> 1, c0 = (t & 1) * 16;
      const float4* src = (const float4*)(x + (size_t)(r0 + r) * EMBED + k0 + c0);
#pragma unroll
      for (int i = 0; i < 4; ++i) {
        float4 f = src[i];
        Xs[r][c0 + 4 * i + 0] = (h16)f.x;
        Xs[r][c0 + 4 * i + 1] = (h16)f.y;
        Xs[r][c0 + 4 * i + 2] = (h16)f.z;
        Xs[r][c0 + 4 * i + 3] = (h16)f.w;
      }
    }
    // stage W chunk [32 x 128] transposed as f16
    {
      const int k = t >> 3, n0 = (t & 7) * 16;
      const float4* src = (const float4*)(W + (size_t)(k0 + k) * DH + n0);
#pragma unroll
      for (int i = 0; i < 4; ++i) {
        float4 f = src[i];
        Wt[n0 + 4 * i + 0][k] = (h16)f.x;
        Wt[n0 + 4 * i + 1][k] = (h16)f.y;
        Wt[n0 + 4 * i + 2][k] = (h16)f.z;
        Wt[n0 + 4 * i + 3][k] = (h16)f.w;
      }
    }
    __syncthreads();

    // hoist all operands, then issue 8 WMMAs back-to-back
    v16h a = make_a(&Xs[16 * w + ln][0], 16 * h);
    v16h bb[8];
#pragma unroll
    for (int n = 0; n < 8; ++n) bb[n] = make_b(&Wt[16 * n + ln][0], 16 * h);
#pragma unroll
    for (int n = 0; n < 8; ++n) acc[n] = wmma_f16(a, bb[n], acc[n]);
    __syncthreads();
  }

  // epilogue: bias add, convert to f16, store
#pragma unroll
  for (int n = 0; n < 8; ++n) {
    const float bias = bv_[16 * n + ln];
#pragma unroll
    for (int i = 0; i < 8; ++i) {
      const int row = r0 + 16 * w + 8 * h + i;
      out[(size_t)row * DH + 16 * n + ln] = (h16)(acc[n][i] + bias);
    }
  }
}

// ============ Kernel 2: causal flash attention over f16 Q/K/V ============
__global__ __launch_bounds__(256) void attn_kernel(const h16* __restrict__ qkv,
                                                   float* __restrict__ out) {
  __shared__ h16 Qs[128][136];       // [q row][dh]
  __shared__ h16 Ks[128][136];       // [k row][dh]
  __shared__ h16 Vt[128][136];       // transposed: [dh][k row]
  __shared__ h16 Ps[8][16][136];     // per-wave P staging [q row][k col]

  const int qt = blockIdx.x;
  const int b  = blockIdx.y;
  const h16* Q = qkv;
  const h16* K = qkv + (size_t)BS * DH;
  const h16* V = qkv + (size_t)2 * BS * DH;

  const int t    = threadIdx.x;
  const int w    = t >> 5;
  const int lane = t & 31;
  const int h    = lane >> 4;
  const int ln   = lane & 15;
  const int r    = t >> 1;
  const int c0   = (t & 1) * 64;

  // async-load Q tile once (completion folded into first iteration's wait)
  {
    const h16* src = Q + ((size_t)(b * SEQ + qt * 128 + r)) * DH + c0;
#pragma unroll
    for (int i = 0; i < 8; ++i) async_cp16(src + 8 * i, &Qs[r][c0 + 8 * i]);
  }

  float m_run[8], l_run[8];
  v8f o[8];
#pragma unroll
  for (int i = 0; i < 8; ++i) { m_run[i] = -__builtin_inff(); l_run[i] = 0.0f; }
#pragma unroll
  for (int n = 0; n < 8; ++n) o[n] = (v8f)(0.0f);

  for (int j = 0; j <= qt; ++j) {
    __syncthreads();  // previous iteration's readers done with Ks/Vt
    const size_t kvbase = ((size_t)(b * SEQ + j * 128 + r)) * DH + c0;
    // K tile: pure copy -> async global->LDS (ASYNCcnt path)
#pragma unroll
    for (int i = 0; i < 8; ++i)
      async_cp16(K + kvbase + 8 * i, &Ks[r][c0 + 8 * i]);
    // V tile: transpose during staging (VGPR path)
    {
      const v8h* src = (const v8h*)(V + kvbase);
#pragma unroll
      for (int i = 0; i < 8; ++i) {
        v8h g = src[i];
#pragma unroll
        for (int u = 0; u < 8; ++u) Vt[c0 + 8 * i + u][r] = g[u];
      }
    }
    if (j < qt) {  // pull next block toward L2 (global_prefetch_b8)
      __builtin_prefetch(K + kvbase + (size_t)128 * DH, 0, 1);
      __builtin_prefetch(V + kvbase + (size_t)128 * DH, 0, 1);
    }
    wait_async0();
    __syncthreads();

    // S = Q @ K^T  (fp32 accum): per K-step, bulk-load operands then 8 WMMAs
    v8f s[8];
#pragma unroll
    for (int n = 0; n < 8; ++n) s[n] = (v8f)(0.0f);
#pragma unroll
    for (int kk = 0; kk < 4; ++kk) {
      const int k0 = kk * 32;
      v16h a = make_a(&Qs[16 * w + ln][0], k0 + 16 * h);
      v16h bb[8];
#pragma unroll
      for (int n = 0; n < 8; ++n) bb[n] = make_b(&Ks[16 * n + ln][0], k0 + 16 * h);
#pragma unroll
      for (int n = 0; n < 8; ++n) s[n] = wmma_f16(a, bb[n], s[n]);
    }

    // scale (reference MULTIPLIES by sqrt(embed_dim) = 32) + causal mask + row max
    const bool diag = (j == qt);
    float mloc[8];
#pragma unroll
    for (int i = 0; i < 8; ++i) mloc[i] = -__builtin_inff();
#pragma unroll
    for (int n = 0; n < 8; ++n) {
#pragma unroll
      for (int i = 0; i < 8; ++i) {
        float v = s[n][i] * 32.0f;
        if (diag) {
          const int qr = w * 16 + h * 8 + i;
          const int kc = n * 16 + ln;
          if (kc > qr) v = -__builtin_inff();
        }
        s[n][i] = v;
        mloc[i] = fmaxf(mloc[i], v);
      }
    }
#pragma unroll
    for (int msk = 1; msk <= 8; msk <<= 1)
#pragma unroll
      for (int i = 0; i < 8; ++i) mloc[i] = fmaxf(mloc[i], __shfl_xor(mloc[i], msk, 32));

    float mnew[8], alpha[8], lloc[8];
#pragma unroll
    for (int i = 0; i < 8; ++i) {
      mnew[i]  = fmaxf(m_run[i], mloc[i]);
      alpha[i] = __expf(m_run[i] - mnew[i]);  // first iter: exp(-inf)=0
      lloc[i]  = 0.0f;
    }
    // P = exp(S - m), stage to LDS as f16 for the PV matmul
#pragma unroll
    for (int n = 0; n < 8; ++n) {
#pragma unroll
      for (int i = 0; i < 8; ++i) {
        float p = __expf(s[n][i] - mnew[i]);
        lloc[i] += p;
        Ps[w][h * 8 + i][n * 16 + ln] = (h16)p;
      }
    }
#pragma unroll
    for (int msk = 1; msk <= 8; msk <<= 1)
#pragma unroll
      for (int i = 0; i < 8; ++i) lloc[i] += __shfl_xor(lloc[i], msk, 32);
#pragma unroll
    for (int i = 0; i < 8; ++i) {
      l_run[i] = l_run[i] * alpha[i] + lloc[i];
      m_run[i] = mnew[i];
    }
#pragma unroll
    for (int n = 0; n < 8; ++n)
#pragma unroll
      for (int i = 0; i < 8; ++i) o[n][i] *= alpha[i];

    __syncthreads();  // Ps visible cross-lane

    // O += P @ V   (contraction over 128 keys, V pre-transposed in LDS)
#pragma unroll
    for (int kk = 0; kk < 4; ++kk) {
      const int k0 = kk * 32;
      v16h a = make_a(&Ps[w][ln][0], k0 + 16 * h);
      v16h bb[8];
#pragma unroll
      for (int n = 0; n < 8; ++n) bb[n] = make_b(&Vt[16 * n + ln][0], k0 + 16 * h);
#pragma unroll
      for (int n = 0; n < 8; ++n) o[n] = wmma_f16(a, bb[n], o[n]);
    }
  }

  // normalize + store fp32
  float rinv[8];
#pragma unroll
  for (int i = 0; i < 8; ++i) rinv[i] = 1.0f / l_run[i];
#pragma unroll
  for (int n = 0; n < 8; ++n) {
#pragma unroll
    for (int i = 0; i < 8; ++i) {
      const int row = qt * 128 + w * 16 + h * 8 + i;
      out[((size_t)b * SEQ + row) * DH + n * 16 + ln] = o[n][i] * rinv[i];
    }
  }
}

extern "C" void kernel_launch(void* const* d_in, const int* in_sizes, int n_in,
                              void* d_out, int out_size, void* d_ws, size_t ws_size,
                              hipStream_t stream) {
  (void)in_sizes; (void)n_in; (void)out_size; (void)ws_size;
  const float* x  = (const float*)d_in[0];
  const float* Wq = (const float*)d_in[1];
  const float* bq = (const float*)d_in[2];
  const float* Wk = (const float*)d_in[3];
  const float* bk = (const float*)d_in[4];
  const float* Wv = (const float*)d_in[5];
  const float* bv = (const float*)d_in[6];
  h16* qkv = (h16*)d_ws;  // Q | K | V, each BS*DH f16 (total ~25 MB)

  qkv_proj_kernel<<<dim3(BS / 128, 3), 256, 0, stream>>>(x, Wq, bq, Wk, bk, Wv, bv, qkv);
  attn_kernel<<<dim3(SEQ / 128, BATCH), 256, 0, stream>>>(qkv, (float*)d_out);
}
